// MIMICLSTMLosModel_41738492183165
// MI455X (gfx1250) — matrix operand order repaired and checked by reading, not compiled, not run on hardware
//
#include <hip/hip_runtime.h>

// ---------------------------------------------------------------------------
// Fused LSTM (B=4096, T=256, F=64, H=128, S=32) for MI455X (gfx1250).
//
// Strategy: one workgroup (8 wave32) per 16-row batch tile; combined weights
// [W_ih|W_hh] -> bf16 [512 x 192] resident in LDS (~200KB of 320KB/WGP).
// Per timestep: gates[16x512] = [x_t|h][16x192] @ W^T via
// v_wmma_f32_16x16x32_bf16 (6 K-chunks x 4 N-tiles per wave), fp32 accum
// initialized with bias; then activations + LSTM elementwise update with
// h/c state kept in LDS. x_{t+1} is staged into LDS during the elementwise
// phase. This avoids materializing px ([T,B,4H] = 2GB) entirely: HBM traffic
// is essentially just x (256MB ~ 11us @ 23.3TB/s) vs ~4GB for the reference.
// ---------------------------------------------------------------------------

typedef __bf16 bf16_t;
typedef __attribute__((ext_vector_type(16))) __bf16 v16bf;
typedef __attribute__((ext_vector_type(8)))  __bf16 v8bf;
typedef __attribute__((ext_vector_type(4)))  __bf16 v4bf;
typedef __attribute__((ext_vector_type(8)))  float  v8f;
typedef __attribute__((ext_vector_type(4)))  float  v4f;

#define T_STEPS  256
#define F_IN     64
#define H_DIM    128
#define S_DIM    32
#define G4       512           // 4*H
#define K_TOT    192           // F + H
#define M_TILE   16            // batch rows per workgroup
// Padded LDS strides (chosen so 16-lane row-strided b128 loads are conflict-free)
#define W_STRIDE 200           // bf16 elems per weight row   (100 dwords, 100%64=36)
#define G_STRIDE 516           // fp32 elems per gates row
#define C_STRIDE 132           // fp32 elems per c row        (132%64=4)
#define H_STRIDE 136           // bf16 elems per h row        (68 dwords, 68%64=4)
#define X_STRIDE 72            // bf16 elems per x row        (36 dwords)

__device__ __forceinline__ float sig_f(float v)  { return 1.0f / (1.0f + __expf(-v)); }
__device__ __forceinline__ float tanh_f(float v) { return 2.0f / (1.0f + __expf(-2.0f * v)) - 1.0f; }

__device__ __forceinline__ v16bf cat16(v8bf lo, v8bf hi) {
  return __builtin_shufflevector(lo, hi, 0,1,2,3,4,5,6,7,8,9,10,11,12,13,14,15);
}

__global__ __launch_bounds__(256, 1)
void lstm_fused_kernel(const float* __restrict__ x,     const float* __restrict__ stat,
                       const float* __restrict__ hs0,   const float* __restrict__ cs0,
                       const float* __restrict__ W_ih,  const float* __restrict__ W_hh,
                       const float* __restrict__ b_ih,  const float* __restrict__ b_hh,
                       const float* __restrict__ W_lin, const float* __restrict__ b_lin,
                       float* __restrict__ out)
{
  // LDS: 204800 + 2048 + 33024 + 8448 + 4352 + 2304 = 254976 B  (< 320KB/WGP)
  __shared__ bf16_t Wlds[G4 * W_STRIDE];        // combined [W_ih|W_hh] bf16
  __shared__ float  biasLds[G4];                // b_ih + b_hh
  __shared__ float  gatesLds[M_TILE * G_STRIDE];// activated gates (also final scratch)
  __shared__ float  cLds[M_TILE * C_STRIDE];    // cell state fp32
  __shared__ bf16_t hLds[M_TILE * H_STRIDE];    // hidden state bf16 (A-operand source)
  __shared__ bf16_t xLds[M_TILE * X_STRIDE];    // staged x_t bf16

  const int tid  = threadIdx.x;
  const int lane = tid & 31;
  const int wave = tid >> 5;        // 0..7
  const int lrow = lane & 15;       // 0..15
  const int lhi  = lane >> 4;       // 0 or 1
  const int b0   = blockIdx.x * M_TILE;

  // ---------------- one-time init ----------------
  // Weights: row g holds [W_ih[g][0:64] | W_hh[g][0:128]] as bf16.
  for (int r = wave; r < G4; r += 8) {
    const float* srcI = W_ih + r * F_IN;
    const float* srcH = W_hh + r * H_DIM;
    for (int k = lane; k < K_TOT; k += 32) {
      float v = (k < F_IN) ? srcI[k] : srcH[k - F_IN];
      Wlds[r * W_STRIDE + k] = (bf16_t)v;
    }
  }
  for (int g = tid; g < G4; g += 256) biasLds[g] = b_ih[g] + b_hh[g];

  {
    // h0 / c0 : 16 rows x 128, 8 elements per thread
    const int row = tid >> 4;
    const int hd0 = (tid & 15) * 8;
    const size_t gb = (size_t)(b0 + row) * H_DIM + hd0;
    v4f h0a = *(const v4f*)(hs0 + gb);
    v4f h0b = *(const v4f*)(hs0 + gb + 4);
    v4f c0a = *(const v4f*)(cs0 + gb);
    v4f c0b = *(const v4f*)(cs0 + gb + 4);
    v8bf hb;
#pragma unroll
    for (int i = 0; i < 4; ++i) { hb[i] = (bf16_t)h0a[i]; hb[4 + i] = (bf16_t)h0b[i]; }
    *(v8bf*)(hLds + row * H_STRIDE + hd0) = hb;
    *(v4f*)(cLds + row * C_STRIDE + hd0)     = c0a;
    *(v4f*)(cLds + row * C_STRIDE + hd0 + 4) = c0b;

    // x_0 : 16 rows x 64, 4 elements per thread
    const int k0 = (tid & 15) * 4;
    v4f xv = *(const v4f*)(x + ((size_t)(b0 + row) * T_STEPS + 0) * F_IN + k0);
    v4bf xb;
#pragma unroll
    for (int i = 0; i < 4; ++i) xb[i] = (bf16_t)xv[i];
    *(v4bf*)(xLds + row * X_STRIDE + k0) = xb;
  }
  __syncthreads();

  const int colBase  = wave * 64;   // this wave's 64 gate columns
  const int gateType = wave >> 1;   // 0:i 1:f 2:g 3:o (PyTorch order)

  // ---------------- time loop ----------------
  for (int t = 0; t < T_STEPS; ++t) {
    // --- A fragments: [x_t | h] 16x192 bf16, 6 chunks of 16x32 ---
    // ISA A layout (16-bit, 16x32): lanes 0-15 M=lane hold K{0..7,16..23};
    // lanes 16-31 hold K{8..15,24..31}.
    v16bf afrag[6];
#pragma unroll
    for (int kc = 0; kc < 2; ++kc) {
      const int koff = kc * 32 + 8 * lhi;
      v8bf lo = *(const v8bf*)(xLds + lrow * X_STRIDE + koff);
      v8bf hi = *(const v8bf*)(xLds + lrow * X_STRIDE + koff + 16);
      afrag[kc] = cat16(lo, hi);
    }
#pragma unroll
    for (int kc = 0; kc < 4; ++kc) {
      const int koff = kc * 32 + 8 * lhi;
      v8bf lo = *(const v8bf*)(hLds + lrow * H_STRIDE + koff);
      v8bf hi = *(const v8bf*)(hLds + lrow * H_STRIDE + koff + 16);
      afrag[2 + kc] = cat16(lo, hi);
    }

    // --- accumulators initialized with bias (broadcast along M) ---
    v8f acc[4];
#pragma unroll
    for (int j = 0; j < 4; ++j) {
      const float bv = biasLds[colBase + j * 16 + lrow];
      acc[j] = (v8f){bv, bv, bv, bv, bv, bv, bv, bv};
    }

    // --- gates += A @ W^T via WMMA; B (K x N) column n = weight row n ---
    // B layout (16-bit, 32x16), matching A: lanes 0-15 N=lane hold K{0..7,16..23},
    // lanes 16-31 hold K{8..15,24..31} from the weight row.
#pragma unroll
    for (int kc = 0; kc < 6; ++kc) {
#pragma unroll
      for (int j = 0; j < 4; ++j) {
        const bf16_t* wrow = Wlds + (colBase + j * 16 + lrow) * W_STRIDE
                                  + kc * 32 + 8 * lhi;
        v8bf lo = *(const v8bf*)(wrow);
        v8bf hi = *(const v8bf*)(wrow + 16);
        v16bf bfrag = cat16(lo, hi);
        acc[j] = __builtin_amdgcn_wmma_f32_16x16x32_bf16(
            false, afrag[kc], false, bfrag, (short)0, acc[j], false, false);
      }
    }

    // --- activation + scatter to gates LDS ---
    // C/D layout: VGPR r, lanes 0-15 -> (M=r, N=lane); lanes 16-31 -> M=r+8.
#pragma unroll
    for (int j = 0; j < 4; ++j) {
      const int col = colBase + j * 16 + lrow;
#pragma unroll
      for (int r = 0; r < 8; ++r) {
        const int row = r + 8 * lhi;
        float v = acc[j][r];
        v = (gateType == 2) ? tanh_f(v) : sig_f(v);
        gatesLds[row * G_STRIDE + col] = v;
      }
    }
    __syncthreads();

    // --- elementwise LSTM update (each thread: 1 row x 8 hidden units) ---
    {
      const int row = tid >> 4;
      const int hd0 = (tid & 15) * 8;
      const float* gp = gatesLds + row * G_STRIDE;
      v4f iv0 = *(const v4f*)(gp + 0 * H_DIM + hd0);
      v4f iv1 = *(const v4f*)(gp + 0 * H_DIM + hd0 + 4);
      v4f fv0 = *(const v4f*)(gp + 1 * H_DIM + hd0);
      v4f fv1 = *(const v4f*)(gp + 1 * H_DIM + hd0 + 4);
      v4f gv0 = *(const v4f*)(gp + 2 * H_DIM + hd0);
      v4f gv1 = *(const v4f*)(gp + 2 * H_DIM + hd0 + 4);
      v4f ov0 = *(const v4f*)(gp + 3 * H_DIM + hd0);
      v4f ov1 = *(const v4f*)(gp + 3 * H_DIM + hd0 + 4);
      v4f c0 = *(const v4f*)(cLds + row * C_STRIDE + hd0);
      v4f c1 = *(const v4f*)(cLds + row * C_STRIDE + hd0 + 4);
      v8bf hb;
#pragma unroll
      for (int q = 0; q < 4; ++q) {
        float cn = fv0[q] * c0[q] + iv0[q] * gv0[q];
        c0[q] = cn;
        hb[q] = (bf16_t)(ov0[q] * tanh_f(cn));
      }
#pragma unroll
      for (int q = 0; q < 4; ++q) {
        float cn = fv1[q] * c1[q] + iv1[q] * gv1[q];
        c1[q] = cn;
        hb[4 + q] = (bf16_t)(ov1[q] * tanh_f(cn));
      }
      *(v4f*)(cLds + row * C_STRIDE + hd0)     = c0;
      *(v4f*)(cLds + row * C_STRIDE + hd0 + 4) = c1;
      *(v8bf*)(hLds + row * H_STRIDE + hd0) = hb;

      // stage x_{t+1} into LDS (bf16); prefetch t+2 toward L2
      const int k0 = (tid & 15) * 4;
      const int tn = (t + 1 < T_STEPS) ? (t + 1) : (T_STEPS - 1);
      const float* xsrc = x + ((size_t)(b0 + row) * T_STEPS + tn) * F_IN + k0;
      v4f xv = *(const v4f*)(xsrc);
      v4bf xb;
#pragma unroll
      for (int i = 0; i < 4; ++i) xb[i] = (bf16_t)xv[i];
      *(v4bf*)(xLds + row * X_STRIDE + k0) = xb;
      if (t + 2 < T_STEPS)
        __builtin_prefetch(xsrc + F_IN, 0, 3);   // global_prefetch_b8
    }
    __syncthreads();
  }

  // ---------------- fused final linear: out = [h|static] @ W_lin^T + b ----
  {
    const int row  = tid >> 4;
    const int part = tid & 15;
    float s = 0.0f;
#pragma unroll
    for (int jj = 0; jj < 10; ++jj) {             // 16 threads x 10 = 160 = H+S
      const int idx = part * 10 + jj;
      const float v = (idx < H_DIM)
          ? (float)hLds[row * H_STRIDE + idx]
          : stat[(size_t)(b0 + row) * S_DIM + (idx - H_DIM)];
      s += v * W_lin[idx];
    }
    gatesLds[row * 16 + part] = s;
  }
  __syncthreads();
  if (tid < M_TILE) {
    float s = b_lin[0];
#pragma unroll
    for (int p = 0; p < 16; ++p) s += gatesLds[tid * 16 + p];
    out[b0 + tid] = s;
  }
}

extern "C" void kernel_launch(void* const* d_in, const int* in_sizes, int n_in,
                              void* d_out, int out_size, void* d_ws, size_t ws_size,
                              hipStream_t stream) {
  const float* x     = (const float*)d_in[0];
  const float* stat  = (const float*)d_in[1];
  const float* hs0   = (const float*)d_in[2];
  const float* cs0   = (const float*)d_in[3];
  const float* W_ih  = (const float*)d_in[4];
  const float* W_hh  = (const float*)d_in[5];
  const float* b_ih  = (const float*)d_in[6];
  const float* b_hh  = (const float*)d_in[7];
  const float* W_lin = (const float*)d_in[8];
  const float* b_lin = (const float*)d_in[9];
  float* out = (float*)d_out;

  const int Bn = in_sizes[2] / H_DIM;           // 4096
  dim3 grid(Bn / M_TILE);                       // 256 workgroups
  lstm_fused_kernel<<<grid, 256, 0, stream>>>(x, stat, hs0, cs0, W_ih, W_hh,
                                              b_ih, b_hh, W_lin, b_lin, out);
}
